// MoEGate_34308198761252
// MI455X (gfx1250) — matrix operand order repaired and checked by reading, not compiled, not run on hardware
//
#include <hip/hip_runtime.h>

typedef __attribute__((ext_vector_type(16))) __bf16 v16bf;
typedef __attribute__((ext_vector_type(8)))  __bf16 v8bf;
typedef __attribute__((ext_vector_type(8)))  float  v8f;

#define N_TOKENS  8192
#define HIDDEN    4096
#define N_EXPERTS 64
#define TOP_K     8

#define KC    32   // K per WMMA step (bf16 16x16x32)
#define BROW  40   // padded bf16 row stride for staged weights (80B, 16B aligned, conflict-free)
#define LROW  68   // padded float row stride for logits (bank-conflict-free)

__global__ __launch_bounds__(256) void moe_gate_kernel(
    const float* __restrict__ h, const float* __restrict__ w,
    float* __restrict__ out_w, int* __restrict__ out_i)
{
    __shared__ __align__(16) __bf16 lds_bhi[N_EXPERTS * BROW];
    __shared__ __align__(16) __bf16 lds_blo[N_EXPERTS * BROW];
    __shared__ __align__(16) float  lds_logit[8 * 16 * LROW];

    const int tid  = threadIdx.x;
    const int wave = tid >> 5;
    const int lane = tid & 31;
    const int l15  = lane & 15;
    const int half = lane >> 4;

    const int tokBase = blockIdx.x * 128 + wave * 16;

    v8f acc0 = {0.f,0.f,0.f,0.f,0.f,0.f,0.f,0.f};
    v8f acc1 = acc0, acc2 = acc0, acc3 = acc0;
    v8f acc[4] = {acc0, acc1, acc2, acc3};

    // Per-lane A addressing: row = l15, K-chunk base depends on half-wave
    const float* hrow = h + (size_t)(tokBase + l15) * HIDDEN + half * 8;
    // Staging assignment: each thread handles 8 contiguous K of one expert row
    const int st_idx = tid * 8;            // 0..2047
    const int st_e   = st_idx >> 5;        // expert 0..63
    const int st_k   = st_idx & 31;        // k offset 0/8/16/24

    for (int kc = 0; kc < HIDDEN; kc += KC) {
        // ---- load A fragment (global, fp32) and split to bf16 hi/lo ----
        const float* ap = hrow + kc;
        float4 a0 = *(const float4*)(ap + 0);
        float4 a1 = *(const float4*)(ap + 4);
        float4 a2 = *(const float4*)(ap + 16);
        float4 a3 = *(const float4*)(ap + 20);
        float af[16] = {a0.x,a0.y,a0.z,a0.w, a1.x,a1.y,a1.z,a1.w,
                        a2.x,a2.y,a2.z,a2.w, a3.x,a3.y,a3.z,a3.w};
        v16bf ahi, alo;
        #pragma unroll
        for (int i = 0; i < 16; ++i) {
            float x  = af[i];
            __bf16 hb = (__bf16)x;
            ahi[i] = hb;
            alo[i] = (__bf16)(x - (float)hb);
        }

        // ---- cooperative stage of weight K-chunk into LDS as bf16 hi/lo ----
        __syncthreads();  // previous iteration's B reads complete
        {
            const float* wp = w + (size_t)st_e * HIDDEN + kc + st_k;
            float4 f0 = *(const float4*)(wp + 0);
            float4 f1 = *(const float4*)(wp + 4);
            float fs[8] = {f0.x,f0.y,f0.z,f0.w, f1.x,f1.y,f1.z,f1.w};
            v8bf hv, lv;
            #pragma unroll
            for (int i = 0; i < 8; ++i) {
                float x  = fs[i];
                __bf16 hb = (__bf16)x;
                hv[i] = hb;
                lv[i] = (__bf16)(x - (float)hb);
            }
            *reinterpret_cast<v8bf*>(&lds_bhi[st_e * BROW + st_k]) = hv;
            *reinterpret_cast<v8bf*>(&lds_blo[st_e * BROW + st_k]) = lv;
        }
        __syncthreads();  // staged B visible to all waves

        // ---- 4 expert tiles: read B from LDS, 3 WMMAs each (bf16x3 split) ----
        #pragma unroll
        for (int t = 0; t < 4; ++t) {
            const int col  = t * 16 + l15;       // expert column (N = lane&15)
            const int koff = half * 8;           // K = 8*half+0..7 and 16+8*half+0..7
            const v8bf* bhp = reinterpret_cast<const v8bf*>(&lds_bhi[col * BROW + koff]);
            const v8bf* blp = reinterpret_cast<const v8bf*>(&lds_blo[col * BROW + koff]);
            v8bf bh0 = bhp[0], bh1 = bhp[2];
            v8bf bl0 = blp[0], bl1 = blp[2];
            v16bf bhi, blo;
            #pragma unroll
            for (int i = 0; i < 8; ++i) {
                bhi[i] = bh0[i]; bhi[i + 8] = bh1[i];
                blo[i] = bl0[i]; blo[i + 8] = bl1[i];
            }
            acc[t] = __builtin_amdgcn_wmma_f32_16x16x32_bf16(
                false, ahi, false, bhi, (short)0, acc[t], false, false);
            acc[t] = __builtin_amdgcn_wmma_f32_16x16x32_bf16(
                false, alo, false, bhi, (short)0, acc[t], false, false);
            acc[t] = __builtin_amdgcn_wmma_f32_16x16x32_bf16(
                false, ahi, false, blo, (short)0, acc[t], false, false);
        }
    }

    // ---- spill logits to LDS: C layout lane<16: N=lane, M=vgpr; lane>=16: M=vgpr+8 ----
    #pragma unroll
    for (int t = 0; t < 4; ++t) {
        #pragma unroll
        for (int r = 0; r < 8; ++r) {
            int tok = r + half * 8;
            int e   = t * 16 + l15;
            lds_logit[(wave * 16 + tok) * LROW + e] = acc[t][r];
        }
    }
    __syncthreads();

    // ---- softmax + top-8 per token (16 active lanes per wave, 1 token each) ----
    if (lane < 16) {
        float* row = &lds_logit[(wave * 16 + lane) * LROW];
        float m = row[0];
        #pragma unroll 4
        for (int e = 1; e < N_EXPERTS; ++e) m = fmaxf(m, row[e]);
        float s = 0.f;
        #pragma unroll 4
        for (int e = 0; e < N_EXPERTS; ++e) {
            float ev = expf(row[e] - m);
            row[e] = ev;
            s += ev;
        }
        float inv = 1.f / s;
        int token = tokBase + lane;
        float* ow = out_w + (size_t)token * TOP_K;
        int*   oi = out_i + (size_t)token * TOP_K;
        for (int j = 0; j < TOP_K; ++j) {
            float bv = -1.f; int bi = 0;
            for (int e = 0; e < N_EXPERTS; ++e) {
                float v = row[e];
                if (v > bv) { bv = v; bi = e; }   // strict '>' => first index on ties
            }
            ow[j] = bv * inv;
            oi[j] = bi;
            row[bi] = -1.f;
        }
    }
}

extern "C" void kernel_launch(void* const* d_in, const int* in_sizes, int n_in,
                              void* d_out, int out_size, void* d_ws, size_t ws_size,
                              hipStream_t stream) {
    const float* h = (const float*)d_in[0];   // [8192, 4096] fp32
    const float* w = (const float*)d_in[1];   // [64, 4096]   fp32
    float* out_w = (float*)d_out;                                   // [8192, 8] fp32
    int*   out_i = (int*)d_out + (size_t)N_TOKENS * TOP_K;          // [8192, 8] int32
    moe_gate_kernel<<<N_TOKENS / 128, 256, 0, stream>>>(h, w, out_w, out_i);
}
